// SpMiddleResNetFHD_79568564125781
// MI455X (gfx1250) — matrix operand order repaired and checked by reading, not compiled, not run on hardware
//
#include <hip/hip_runtime.h>
#include <hip/hip_bf16.h>
#include <stdint.h>

typedef _Float16 v16h __attribute__((ext_vector_type(16)));
typedef _Float16 v8h  __attribute__((ext_vector_type(8)));
typedef float    v8f  __attribute__((ext_vector_type(8)));

// ---------------------------------------------------------------------------
// Small helper kernels
// ---------------------------------------------------------------------------

__global__ void zero_dwords_kernel(uint32_t* p, long long n) {
  long long i = (long long)blockIdx.x * blockDim.x + threadIdx.x;
  if (i < n) p[i] = 0u;
}

// Scatter voxel features (NVOX x 5 f32) into dense f16 [z][y][x][16] + byte mask.
__global__ void densify_kernel(const float* __restrict__ vf, const int* __restrict__ coors,
                               _Float16* __restrict__ x, unsigned char* __restrict__ mask, int n) {
  int i = blockIdx.x * blockDim.x + threadIdx.x;
  if (i >= n) return;
  int z = coors[i * 4 + 1], y = coors[i * 4 + 2], xx = coors[i * 4 + 3];
  long long p = ((long long)z * 128 + y) * 128 + xx;
  mask[p] = 1;
#pragma unroll
  for (int c = 0; c < 5; ++c) x[p * 16 + c] = (_Float16)vf[i * 5 + c];
}

// scale = g*rsqrt(v+eps); shift = b - m*scale (+ conv_bias*scale folded in)
__global__ void prep_bn_kernel(const float* g, const float* b, const float* m, const float* v,
                               const float* bias, float* scale, float* shift, int C) {
  int c = blockIdx.x * blockDim.x + threadIdx.x;
  if (c >= C) return;
  float s = g[c] * rsqrtf(v[c] + 1e-3f);
  float sh = b[c] - m[c] * s;
  if (bias) sh += bias[c] * s;
  scale[c] = s;
  shift[c] = sh;
}

// OIDHW f32 -> [tap][ci(pad to Cin16)][co] f16  (K-major x N for the GEMM B matrix)
__global__ void repack_w_kernel(const float* __restrict__ w, _Float16* __restrict__ wpk,
                                int Cin, int Cin16, int Cout, int KD, int KH, int KW) {
  int total = KD * KH * KW * Cin16 * Cout;
  int idx = blockIdx.x * blockDim.x + threadIdx.x;
  if (idx >= total) return;
  int co = idx % Cout;
  int t = idx / Cout;
  int ci = t % Cin16;
  int tap = t / Cin16;
  int kw = tap % KW; int t2 = tap / KW;
  int kh = t2 % KH; int kd = t2 / KH;
  float val = 0.f;
  if (ci < Cin) val = w[(((long long)co * Cin + ci) * KD + kd) * KH * KW + kh * KW + kw];
  wpk[idx] = (_Float16)val;
}

// new_mask[out] = any(input mask in receptive field)
__global__ void mask_down_kernel(const unsigned char* __restrict__ mi, unsigned char* __restrict__ mo,
                                 int D, int H, int W, int OD, int OH, int OW,
                                 int KD, int KH, int KW, int sd, int sh, int sw,
                                 int pd, int ph, int pw) {
  int P = OD * OH * OW;
  int idx = blockIdx.x * blockDim.x + threadIdx.x;
  if (idx >= P) return;
  int ox = idx % OW; int t = idx / OW;
  int oy = t % OH;   int oz = t / OH;
  unsigned char any = 0;
  for (int kd = 0; kd < KD; ++kd)
    for (int kh = 0; kh < KH; ++kh)
      for (int kw = 0; kw < KW; ++kw) {
        int iz = oz * sd - pd + kd, iy = oy * sh - ph + kh, ix = ox * sw - pw + kw;
        if ((unsigned)iz < (unsigned)D && (unsigned)iy < (unsigned)H && (unsigned)ix < (unsigned)W)
          any |= mi[((long long)iz * H + iy) * W + ix];
      }
  mo[idx] = any ? 1 : 0;
}

// f16 [z][y][x][C] -> f32 NC(D)HW flattened as ((c*D+z)*H+y)*W+x
__global__ void to_nchw_kernel(const _Float16* __restrict__ src, float* __restrict__ dst,
                               int C, int D, int H, int W) {
  int total = C * D * H * W;
  int idx = blockIdx.x * blockDim.x + threadIdx.x;
  if (idx >= total) return;
  int x = idx % W; int t = idx / W;
  int y = t % H;   t /= H;
  int z = t % D;   int c = t / D;
  dst[idx] = (float)src[(((long long)z * H + y) * W + x) * C + c];
}

// ---------------------------------------------------------------------------
// Fused implicit-GEMM 3D conv with WMMA f16->f32.
//   Wave tile: M = 16 output voxels, N = 16*NT output channels.
//   K = Cin16 * KD*KHW*KHW, chunked by 64 (two 32-wide WMMA sub-chunks).
//   KHW (=KH=KW, 3 or 1) is a template param so tap decomposition uses
//   compile-time divisors (no runtime-divide rcp sequences in the loop).
//   A tile: per lane, two 16-K segments; each is one tap + 16 contiguous
//   channels -> two aligned v8h global loads (Cin16, k0 multiples of 16).
//   B tile staged transposed [nt][col][k] so fragments are two v8h LDS loads.
//   Epilogue: v = acc*scale[co] + shift[co] (+residual) -> lrelu -> * mask.
// Block = 256 threads = 8 waves; each wave owns one M-tile; B shared in LDS.
// ---------------------------------------------------------------------------
template <int NT, int KHW>
__global__ __launch_bounds__(256) void conv_wmma_kernel(
    const _Float16* __restrict__ x, const _Float16* __restrict__ wpk,
    const float* __restrict__ scale, const float* __restrict__ shift,
    const unsigned char* __restrict__ mask, const _Float16* __restrict__ residual,
    _Float16* __restrict__ y,
    int D, int H, int W, int OD, int OH, int OW,
    int Cin16, int Cout, int KD,
    int sd, int shd, int sw, int pd, int ph, int pw, int lc) {
  __shared__ __attribute__((aligned(16))) _Float16 As[8][16][64];
  __shared__ __attribute__((aligned(16))) _Float16 Bs[NT][16][72];  // [nt][col][k], padded

  const int lane = threadIdx.x & 31;
  const int wave = threadIdx.x >> 5;
  const int P = OD * OH * OW;
  const int HW = OH * OW;
  const int posBase = ((int)blockIdx.x * 8 + wave) * 16;
  const int coBase = (int)blockIdx.y * (16 * NT);
  const int Ktot = KD * KHW * KHW * Cin16;
  const int cmask = Cin16 - 1;

  // Per-lane A-staging coordinates (row m = lane>>1, base k-seg = (lane&1)*16)
  const int m = lane >> 1;
  const int kkBase = (lane & 1) * 16;
  const int posA = posBase + m;
  const bool posOK = posA < P;
  int bz = 0, by = 0, bx = 0;
  if (posOK) {
    int oz = posA / HW; int r = posA - oz * HW;
    int oy = r / OW;    int ox = r - oy * OW;
    bz = oz * sd - pd; by = oy * shd - ph; bx = ox * sw - pw;
  }

  v8f acc[NT];
#pragma unroll
  for (int nt = 0; nt < NT; ++nt) acc[nt] = (v8f){};

  for (int k0 = 0; k0 < Ktot; k0 += 64) {
    // ---- stage A tile (16 x 64): two segments/lane, each one tap ----
#pragma unroll
    for (int h = 0; h < 2; ++h) {
      int kk = kkBase + h * 32;
      int k = k0 + kk;
      v8h va0 = {}, va1 = {};
      if (k < Ktot && posOK) {
        int tap = k >> lc;
        int ciB = k & cmask;
        int kw = tap % KHW;              // compile-time divisor
        int t2 = tap / KHW;
        int kh = t2 % KHW;
        int kd = t2 / KHW;
        int iz = bz + kd, iy = by + kh, ix = bx + kw;
        if ((unsigned)iz < (unsigned)D && (unsigned)iy < (unsigned)H && (unsigned)ix < (unsigned)W) {
          const v8h* px = (const v8h*)(x + (((long long)iz * H + iy) * W + ix) * Cin16 + ciB);
          va0 = px[0];
          va1 = px[1];
        }
      }
      *(v8h*)&As[wave][m][kk]     = va0;
      *(v8h*)&As[wave][m][kk + 8] = va1;
    }
    // ---- stage B tile (64 x 16*NT) transposed into [nt][col][k] ----
    {
      const int nVec = 2 * NT;  // v8h vectors per k-row
      for (int vv = (int)threadIdx.x; vv < 64 * nVec; vv += 256) {
        int kk = vv / nVec, v = vv - kk * nVec;  // nVec is a power of two
        int k = k0 + kk;
        v8h w8 = {};
        if (k < Ktot) w8 = *(const v8h*)(wpk + (long long)k * Cout + coBase + v * 8);
        int nt = v >> 1, n0 = (v & 1) * 8;
#pragma unroll
        for (int j = 0; j < 8; ++j) Bs[nt][n0 + j][kk] = w8[j];
      }
    }
    __builtin_prefetch(wpk + (long long)(k0 + 64) * Cout + coBase, 0, 1);
    __syncthreads();

    // ---- fragments per ISA 7.12.2 layouts; 2 x NT WMMAs per chunk ----
    const int M = lane & 15;
    const int hi = lane >> 4;
    const int ka = hi ? 8 : 0;    // A: lanes0-15 K{0..7,16..23}; lanes16-31 K{8..15,24..31}
    const int kb = hi ? 24 : 16;

#pragma unroll
    for (int h = 0; h < 2; ++h) {
      const int ko = h * 32;
      v8h a0 = *(const v8h*)&As[wave][M][ko + ka];
      v8h a1 = *(const v8h*)&As[wave][M][ko + kb];
      v16h a = __builtin_shufflevector(a0, a1, 0, 1, 2, 3, 4, 5, 6, 7, 8, 9, 10, 11, 12, 13, 14, 15);
#pragma unroll
      for (int nt = 0; nt < NT; ++nt) {
        v8h b0 = *(const v8h*)&Bs[nt][M][ko + ka];
        v8h b1 = *(const v8h*)&Bs[nt][M][ko + kb];
        v16h b = __builtin_shufflevector(b0, b1, 0, 1, 2, 3, 4, 5, 6, 7, 8, 9, 10, 11, 12, 13, 14, 15);
        acc[nt] = __builtin_amdgcn_wmma_f32_16x16x32_f16(false, a, false, b, (short)0, acc[nt],
                                                         false, false);
      }
    }
    __syncthreads();
  }

  // ---- epilogue: BN + (residual) + LeakyReLU + mask ----
  const int N = lane & 15;
  const int mOff = (lane >> 4) * 8;  // D layout: VGPR r -> M=r (lanes0-15), M=8+r (lanes16-31)
#pragma unroll
  for (int nt = 0; nt < NT; ++nt) {
    const int co = coBase + nt * 16 + N;
    const float sc = scale[co], sv = shift[co];
#pragma unroll
    for (int r = 0; r < 8; ++r) {
      int pos = posBase + mOff + r;
      if (pos < P) {
        float v = acc[nt][r] * sc + sv;
        if (residual) v += (float)residual[(long long)pos * Cout + co];
        v = v > 0.f ? v : 0.01f * v;
        v *= (float)mask[pos];
        y[(long long)pos * Cout + co] = (_Float16)v;
      }
    }
  }
}

// ---------------------------------------------------------------------------
// Orchestration
// ---------------------------------------------------------------------------
extern "C" void kernel_launch(void* const* d_in, const int* in_sizes, int n_in,
                              void* d_out, int out_size, void* d_ws, size_t ws_size,
                              hipStream_t stream) {
  (void)in_sizes; (void)n_in; (void)out_size; (void)ws_size;

  const float* vf = (const float*)d_in[0];
  const int* coors = (const int*)d_in[1];

  int pi = 3;
  auto f = [&](int i) { return (const float*)d_in[i]; };
  const float* w_in = f(pi++);
  const float *bnin_g = f(pi++), *bnin_b = f(pi++), *bnin_m = f(pi++), *bnin_v = f(pi++);
  struct Blk { const float *w1,*b1,*g1,*bb1,*m1,*v1,*w2,*b2,*g2,*bb2,*m2,*v2; };
  auto readBlk = [&]() {
    Blk k;
    k.w1 = f(pi++); k.b1 = f(pi++); k.g1 = f(pi++); k.bb1 = f(pi++); k.m1 = f(pi++); k.v1 = f(pi++);
    k.w2 = f(pi++); k.b2 = f(pi++); k.g2 = f(pi++); k.bb2 = f(pi++); k.m2 = f(pi++); k.v2 = f(pi++);
    return k;
  };
  Blk c1_0 = readBlk(), c1_1 = readBlk();
  const float* w2d = f(pi++);
  const float *bn2_g = f(pi++), *bn2_b = f(pi++), *bn2_m = f(pi++), *bn2_v = f(pi++);
  Blk c2_0 = readBlk(), c2_1 = readBlk();
  const float* w3d = f(pi++);
  const float *bn3_g = f(pi++), *bn3_b = f(pi++), *bn3_m = f(pi++), *bn3_v = f(pi++);
  Blk c3_0 = readBlk(), c3_1 = readBlk();
  const float* w4d = f(pi++);
  const float *bn4_g = f(pi++), *bn4_b = f(pi++), *bn4_m = f(pi++), *bn4_v = f(pi++);
  Blk c4_0 = readBlk(), c4_1 = readBlk();
  const float* wex = f(pi++);
  const float *bnex_g = f(pi++), *bnex_b = f(pi++), *bnex_m = f(pi++), *bnex_v = f(pi++);

  // ---- workspace carve ----
  const long long P1 = 41LL * 128 * 128;  // 671744
  const long long P2 = 21LL * 64 * 64;    // 86016
  const long long P3 = 11LL * 32 * 32;    // 11264
  const long long P4 = 5LL * 16 * 16;     // 1280
  const long long P5 = 2LL * 16 * 16;     // 512
  const size_t BUF_BYTES = (size_t)P1 * 16 * sizeof(_Float16);  // 21,495,808

  char* ws = (char*)d_ws;
  size_t off = 0;
  auto take = [&](size_t bytes) {
    void* p = ws + off;
    off += (bytes + 255) & ~(size_t)255;
    return p;
  };
  _Float16* A_  = (_Float16*)take(BUF_BYTES);
  _Float16* B_  = (_Float16*)take(BUF_BYTES);
  _Float16* Cb  = (_Float16*)take(BUF_BYTES);
  _Float16* wpk = (_Float16*)take(1 << 20);
  float* scale  = (float*)take(512);
  float* shift  = (float*)take(512);
  unsigned char* m1 = (unsigned char*)take((size_t)P1);
  unsigned char* m2 = (unsigned char*)take((size_t)P2);
  unsigned char* m3 = (unsigned char*)take((size_t)P3);
  unsigned char* m4 = (unsigned char*)take((size_t)P4);
  unsigned char* m5 = (unsigned char*)take((size_t)P5);

  auto prep = [&](const float* g, const float* b, const float* m, const float* v,
                  const float* bias, int C) {
    prep_bn_kernel<<<1, 128, 0, stream>>>(g, b, m, v, bias, scale, shift, C);
  };
  auto pack = [&](const float* w, int Cin, int Cin16, int Cout, int KD, int KH, int KW) {
    int total = KD * KH * KW * Cin16 * Cout;
    repack_w_kernel<<<(total + 255) / 256, 256, 0, stream>>>(w, wpk, Cin, Cin16, Cout, KD, KH, KW);
  };
  // KH==KW is 3 or 1 for every conv in this network.
  auto conv = [&](const _Float16* x, _Float16* y, const unsigned char* msk, const _Float16* res,
                  int D, int H, int W, int OD, int OH, int OW, int Cin16, int Cout,
                  int KD, int KHW,
                  int sd, int sh, int sw, int pd, int ph, int pw) {
    int P = OD * OH * OW;
    int lc = 31 - __builtin_clz((unsigned)Cin16);  // log2(Cin16), power of two
    dim3 grid((P + 127) / 128, (Cout >= 64) ? Cout / 64 : 1);
    if (KHW == 3) {
      if (Cout >= 64)
        conv_wmma_kernel<4, 3><<<grid, 256, 0, stream>>>(x, wpk, scale, shift, msk, res, y,
                                                         D, H, W, OD, OH, OW, Cin16, Cout, KD,
                                                         sd, sh, sw, pd, ph, pw, lc);
      else if (Cout == 32)
        conv_wmma_kernel<2, 3><<<grid, 256, 0, stream>>>(x, wpk, scale, shift, msk, res, y,
                                                         D, H, W, OD, OH, OW, Cin16, Cout, KD,
                                                         sd, sh, sw, pd, ph, pw, lc);
      else
        conv_wmma_kernel<1, 3><<<grid, 256, 0, stream>>>(x, wpk, scale, shift, msk, res, y,
                                                         D, H, W, OD, OH, OW, Cin16, Cout, KD,
                                                         sd, sh, sw, pd, ph, pw, lc);
    } else {
      conv_wmma_kernel<4, 1><<<grid, 256, 0, stream>>>(x, wpk, scale, shift, msk, res, y,
                                                       D, H, W, OD, OH, OW, Cin16, Cout, KD,
                                                       sd, sh, sw, pd, ph, pw, lc);
    }
  };
  auto blockrun = [&](const Blk& k, int C, const _Float16* xin, _Float16* tmp, _Float16* yout,
                      const unsigned char* msk, int D, int H, int W) {
    prep(k.g1, k.bb1, k.m1, k.v1, k.b1, C);
    pack(k.w1, C, C, C, 3, 3, 3);
    conv(xin, tmp, msk, nullptr, D, H, W, D, H, W, C, C, 3, 3, 1, 1, 1, 1, 1, 1);
    prep(k.g2, k.bb2, k.m2, k.v2, k.b2, C);
    pack(k.w2, C, C, C, 3, 3, 3);
    conv(tmp, yout, msk, xin, D, H, W, D, H, W, C, C, 3, 3, 1, 1, 1, 1, 1, 1);
  };

  // ---- densify (zero stage-1 buffer + mask first; ws is poisoned) ----
  {
    long long nA = (long long)(BUF_BYTES / 4);
    zero_dwords_kernel<<<(unsigned)((nA + 255) / 256), 256, 0, stream>>>((uint32_t*)A_, nA);
    long long nM = P1 / 4;
    zero_dwords_kernel<<<(unsigned)((nM + 255) / 256), 256, 0, stream>>>((uint32_t*)m1, nM);
  }
  densify_kernel<<<(30000 + 255) / 256, 256, 0, stream>>>(vf, coors, A_, m1, 30000);

  // ---- conv_input: 5->16 subm ----
  prep(bnin_g, bnin_b, bnin_m, bnin_v, nullptr, 16);
  pack(w_in, 5, 16, 16, 3, 3, 3);
  conv(A_, B_, m1, nullptr, 41, 128, 128, 41, 128, 128, 16, 16, 3, 3, 1, 1, 1, 1, 1, 1);
  blockrun(c1_0, 16, B_, Cb, A_, m1, 41, 128, 128);   // x = A
  blockrun(c1_1, 16, A_, Cb, B_, m1, 41, 128, 128);   // x = B

  // ---- down2: 16->32 s2 p1 ----
  mask_down_kernel<<<(unsigned)((P2 + 255) / 256), 256, 0, stream>>>(
      m1, m2, 41, 128, 128, 21, 64, 64, 3, 3, 3, 2, 2, 2, 1, 1, 1);
  prep(bn2_g, bn2_b, bn2_m, bn2_v, nullptr, 32);
  pack(w2d, 16, 16, 32, 3, 3, 3);
  conv(B_, A_, m2, nullptr, 41, 128, 128, 21, 64, 64, 16, 32, 3, 3, 2, 2, 2, 1, 1, 1);
  blockrun(c2_0, 32, A_, Cb, B_, m2, 21, 64, 64);     // x = B
  blockrun(c2_1, 32, B_, Cb, A_, m2, 21, 64, 64);     // x = A

  // ---- down3: 32->64 s2 p1 ----
  mask_down_kernel<<<(unsigned)((P3 + 255) / 256), 256, 0, stream>>>(
      m2, m3, 21, 64, 64, 11, 32, 32, 3, 3, 3, 2, 2, 2, 1, 1, 1);
  prep(bn3_g, bn3_b, bn3_m, bn3_v, nullptr, 64);
  pack(w3d, 32, 32, 64, 3, 3, 3);
  conv(A_, B_, m3, nullptr, 21, 64, 64, 11, 32, 32, 32, 64, 3, 3, 2, 2, 2, 1, 1, 1);
  blockrun(c3_0, 64, B_, Cb, A_, m3, 11, 32, 32);     // x = A
  blockrun(c3_1, 64, A_, Cb, B_, m3, 11, 32, 32);     // x = B  (== x3, keep!)

  // ---- down4: 64->128 s2 pad (0,1,1) ----
  mask_down_kernel<<<(unsigned)((P4 + 255) / 256), 256, 0, stream>>>(
      m3, m4, 11, 32, 32, 5, 16, 16, 3, 3, 3, 2, 2, 2, 0, 1, 1);
  prep(bn4_g, bn4_b, bn4_m, bn4_v, nullptr, 128);
  pack(w4d, 64, 64, 128, 3, 3, 3);
  conv(B_, A_, m4, nullptr, 11, 32, 32, 5, 16, 16, 64, 128, 3, 3, 2, 2, 2, 0, 1, 1);
  blockrun(c4_0, 128, A_, Cb, A_, m4, 5, 16, 16);     // in-place residual, B untouched
  blockrun(c4_1, 128, A_, Cb, A_, m4, 5, 16, 16);

  // ---- extra_conv: 128->128 kernel (3,1,1) stride (2,1,1) no pad ----
  mask_down_kernel<<<(unsigned)((P5 + 255) / 256), 256, 0, stream>>>(
      m4, m5, 5, 16, 16, 2, 16, 16, 3, 1, 1, 2, 1, 1, 0, 0, 0);
  prep(bnex_g, bnex_b, bnex_m, bnex_v, nullptr, 128);
  pack(wex, 128, 128, 128, 3, 1, 1);
  conv(A_, Cb, m5, nullptr, 5, 16, 16, 2, 16, 16, 128, 128, 3, 1, 2, 1, 1, 0, 0, 0);

  // ---- outputs: ret (1, 128*2, 16, 16) then second (1, 64*11, 32, 32) ----
  {
    int tot1 = 128 * 2 * 16 * 16;   // 65536
    to_nchw_kernel<<<(tot1 + 255) / 256, 256, 0, stream>>>(Cb, (float*)d_out, 128, 2, 16, 16);
    int tot2 = 64 * 11 * 32 * 32;   // 720896
    to_nchw_kernel<<<(tot2 + 255) / 256, 256, 0, stream>>>(B_, (float*)d_out + tot1, 64, 11, 32, 32);
  }
}
